// TNNColumnLayer_4423816315176
// MI455X (gfx1250) — matrix operand order repaired and checked by reading, not compiled, not run on hardware
//
#include <hip/hip_runtime.h>

#define P 12
#define WMAXC 7
#define THETA 30.0f
#define TPW 4   // tiles (of 32 neurons) per wave

typedef _Float16 v16h __attribute__((ext_vector_type(16)));
typedef float    v8f  __attribute__((ext_vector_type(8)));
typedef float    f4   __attribute__((ext_vector_type(4)));
typedef int      v8i  __attribute__((ext_vector_type(8)));

// Workspace layout (dwords):
//  [0..11]   tiled input spiketimes (the identical row of `inp`)
//  [12..23]  sorted spike times
//  [24..35]  pos[j] = sorted position of synapse j (inverse permutation)
//  [36]      final winning ec (minv)   [37] final pot   [38] winner row (int)
//  [64..]    per-block WTA partials: ec[nb], pot[nb], idx[nb]

// ---------------------------------------------------------------- prep ----
__global__ void tnn_prep(const float* __restrict__ data,
                         float* __restrict__ wsF, unsigned* __restrict__ wsU) {
  if (threadIdx.x == 0 && blockIdx.x == 0) {
    float d[4];
    for (int i = 0; i < 4; ++i) d[i] = data[i];
    int rank[4];
    for (int b = 0; b < 4; ++b) {
      int rk = 0;
      for (int i = 0; i < 4; ++i)
        rk += (d[i] < d[b]) || (d[i] == d[b] && i < b);
      rank[b] = rk;
    }
    for (int j = 0; j < P; ++j) wsF[j] = d[j & 3];   // tiled pattern
    for (int j = 0; j < P; ++j) {
      int b = j & 3, rep = j >> 2;
      int p = 3 * rank[b] + rep;
      wsF[12 + p] = d[b];        // sorted spike times
      wsU[24 + j] = (unsigned)p; // pos[j]
    }
  }
}

// ------------------------------------------------------------------ ec ----
// One wave handles 32 neurons/tile via TWO WMMAs sharing one constant A:
//   A[p,K] = (pos[K] <= p)  (sort permutation folded into triangular mask)
//   B1: weights of rows base..base+15 (lanes 0-15), B2: rows base+16..base+31
//       (moved to lanes 0-15 with 6 packed-f16 shfl_xor(16))
//   D[p,m] = cpot[p] of neuron m; non-decreasing -> crossing = count(<theta),
//   potential = min(>=theta). All reduction updates are branchless (neutral
//   element = +INF) to avoid saveexec divergence in the hot loop.
__global__ __launch_bounds__(256) void tnn_ec(const float* __restrict__ weights,
                                              const float* __restrict__ wsF,
                                              const unsigned* __restrict__ wsU,
                                              float* __restrict__ partEc,
                                              float* __restrict__ partPot,
                                              int* __restrict__ partIdx) {
  __shared__ float sEc[8], sPot[8];
  __shared__ int sIdx[8];
  const int tid  = threadIdx.x;
  const int lane = tid & 31;
  const int h    = lane >> 4;
  const int m    = lane & 15;
  const int wave = tid >> 5;
  const float INF = __builtin_inff();

  // uniform sort data (scalar cached)
  float tiled[P];
  unsigned pos[P];
#pragma unroll
  for (int e = 0; e < P; ++e) tiled[e] = wsF[e];
#pragma unroll
  for (int e = 0; e < P; ++e) pos[e] = wsU[24 + e];

  // constant A, built once (16x32 f16 A-layout: K=16h+e)
  v16h a;
#pragma unroll
  for (int e = 0; e < 16; ++e) {
    int K = e + 16 * h;
    bool on = (K < P) && (pos[K] <= (unsigned)m);
    a[e] = on ? (_Float16)1.0f : (_Float16)0.0f;
  }
  const int bmask = h ? 0 : -1;   // B rows K>=16 (upper-half lanes) are zero
  const v8f czero = {};

  float bEc = INF, bPot = -INF;
  int bIdx = 0x7FFFFFFF;

  const int waveTile0 = (blockIdx.x * 8 + wave) * TPW;
#pragma unroll 1
  for (int tt = 0; tt < TPW; ++tt) {
    const int row = (waveTile0 + tt) * 32 + lane;
    const f4* wp = (const f4*)(weights + (size_t)row * P);
    __builtin_prefetch(weights + (size_t)(row + 32) * P, 0, 0);
    f4 w0 = __builtin_nontemporal_load(wp + 0);
    f4 w1 = __builtin_nontemporal_load(wp + 1);
    f4 w2 = __builtin_nontemporal_load(wp + 2);
    float w[P] = {w0[0], w0[1], w0[2], w0[3],
                  w1[0], w1[1], w1[2], w1[3],
                  w2[0], w2[1], w2[2], w2[3]};

    // pack this lane's weights to f16 pairs
    v16h wh = {};
#pragma unroll
    for (int e = 0; e < P; ++e) wh[e] = (_Float16)w[e];
    v8i pw = __builtin_bit_cast(v8i, wh);
    v8i pw2 = {};
#pragma unroll
    for (int i = 0; i < 6; ++i) pw2[i] = __shfl_xor(pw[i], 16);
    v8i b1i, b2i;
#pragma unroll
    for (int i = 0; i < 8; ++i) { b1i[i] = pw[i] & bmask; b2i[i] = pw2[i] & bmask; }
    v16h b1 = __builtin_bit_cast(v16h, b1i);
    v16h b2 = __builtin_bit_cast(v16h, b2i);

    v8f d1 = __builtin_amdgcn_wmma_f32_16x16x32_f16(false, a, false, b1,
                                                    (short)0, czero, false, false);
    v8f d2 = __builtin_amdgcn_wmma_f32_16x16x32_f16(false, a, false, b2,
                                                    (short)0, czero, false, false);

    // branchless crossing counts / min-of-crossing for both neuron groups
    int c1 = 0, c2 = 0;
    float m1 = INF, m2 = INF;
#pragma unroll
    for (int v = 0; v < 8; ++v) {
      bool valid = (v < 4) || (h == 0);
      float x1 = valid ? d1[v] : INF;   // INF is neutral for both count & min
      float x2 = valid ? d2[v] : INF;
      c1 += (x1 < THETA) ? 1 : 0;
      c2 += (x2 < THETA) ? 1 : 0;
      m1 = fminf(m1, (x1 >= THETA) ? x1 : INF);
      m2 = fminf(m2, (x2 >= THETA) ? x2 : INF);
    }
    float d10 = d1[0], d20 = d2[0];
    int   pc1 = __shfl_xor(c1, 16),  pc2 = __shfl_xor(c2, 16);
    float pm1 = __shfl_xor(m1, 16),  pm2 = __shfl_xor(m2, 16);
    float pd20 = __shfl_xor(d20, 16);

    // lane h==0 -> neuron base+m (group 1); lane h==1 -> neuron base+16+m
    int   cntT   = h ? (c2 + pc2) : (c1 + pc1);
    float potMin = h ? fminf(m2, pm2) : fminf(m1, pm1);
    float cpot0  = h ? pd20 : d10;           // cpot[0] lives on the h==0 lane

    bool  isnull = cntT >= P;
    int   idx = isnull ? 0 : cntT;
    float pot = isnull ? cpot0 : potMin;
    float ec  = isnull ? INF : wsF[12 + idx];

    // RNL refinement (order-invariant): resp = med3(ae+t, 0, w), w >= 0
    float ae[P];
#pragma unroll
    for (int e = 0; e < P; ++e) ae[e] = ec + 1.0f - tiled[e];
    int cnt2 = 0;
    float pmin2 = INF, s0 = 0.0f;
#pragma unroll
    for (int t = 0; t < WMAXC; ++t) {
      float s = 0.0f;
#pragma unroll
      for (int e = 0; e < P; ++e)
        s += __builtin_amdgcn_fmed3f(ae[e] + (float)t, 0.0f, w[e]);
      if (t == 0) s0 = s;                       // compile-time (unrolled)
      cnt2 += (s < THETA) ? 1 : 0;
      pmin2 = fminf(pmin2, (s >= THETA) ? s : INF);
    }
    bool f2 = cnt2 < WMAXC;
    float ecR  = ec + (float)(f2 ? cnt2 : 0);
    float potR = f2 ? pmin2 : s0;
    ec  = isnull ? ec  : ecR;
    pot = isnull ? pot : potR;

    // running per-lane best (min ec, tie -> max pot, tie -> min row)
    bool take = (ec < bEc) ||
                (ec == bEc && (pot > bPot || (pot == bPot && row < bIdx)));
    bEc  = take ? ec  : bEc;
    bPot = take ? pot : bPot;
    bIdx = take ? row : bIdx;
  }

  // wave reduction over 32 distinct neurons
#pragma unroll
  for (int off = 1; off < 32; off <<= 1) {
    float oEc  = __shfl_xor(bEc, off);
    float oPot = __shfl_xor(bPot, off);
    int   oIdx = __shfl_xor(bIdx, off);
    bool take = (oEc < bEc) ||
                (oEc == bEc && (oPot > bPot || (oPot == bPot && oIdx < bIdx)));
    bEc  = take ? oEc  : bEc;
    bPot = take ? oPot : bPot;
    bIdx = take ? oIdx : bIdx;
  }
  if (lane == 0) { sEc[wave] = bEc; sPot[wave] = bPot; sIdx[wave] = bIdx; }
  __syncthreads();
  if (tid == 0) {
    float rEc = sEc[0], rPot = sPot[0];
    int rIdx = sIdx[0];
    for (int i = 1; i < 8; ++i) {
      float e = sEc[i], p = sPot[i];
      int x = sIdx[i];
      bool take = (e < rEc) ||
                  (e == rEc && (p > rPot || (p == rPot && x < rIdx)));
      rEc  = take ? e : rEc;
      rPot = take ? p : rPot;
      rIdx = take ? x : rIdx;
    }
    partEc[blockIdx.x]  = rEc;
    partPot[blockIdx.x] = rPot;
    partIdx[blockIdx.x] = rIdx;
  }
}

// ----------------------------------------------------------------- wta ----
__global__ __launch_bounds__(256) void tnn_wta(const float* __restrict__ partEc,
                                               const float* __restrict__ partPot,
                                               const int* __restrict__ partIdx,
                                               int nPart,
                                               float* __restrict__ wsF,
                                               int* __restrict__ wsI) {
  __shared__ float sEc[256], sPot[256];
  __shared__ int sIdx[256];
  int tid = threadIdx.x;
  float bEc = __builtin_inff(), bPot = -__builtin_inff();
  int bIdx = 0x7FFFFFFF;
  for (int i = tid; i < nPart; i += 256) {
    float e = partEc[i], p = partPot[i];
    int x = partIdx[i];
    bool take = (e < bEc) || (e == bEc && (p > bPot || (p == bPot && x < bIdx)));
    bEc  = take ? e : bEc;
    bPot = take ? p : bPot;
    bIdx = take ? x : bIdx;
  }
  sEc[tid] = bEc; sPot[tid] = bPot; sIdx[tid] = bIdx;
  __syncthreads();
  for (int s = 128; s > 0; s >>= 1) {
    if (tid < s) {
      float eA = sEc[tid],  eB = sEc[tid + s];
      float pA = sPot[tid], pB = sPot[tid + s];
      int   iA = sIdx[tid], iB = sIdx[tid + s];
      bool takeB = (eB < eA) || (eB == eA && (pB > pA || (pB == pA && iB < iA)));
      if (takeB) { sEc[tid] = eB; sPot[tid] = pB; sIdx[tid] = iB; }
    }
    __syncthreads();
  }
  if (tid == 0) { wsF[36] = sEc[0]; wsF[37] = sPot[0]; wsI[38] = sIdx[0]; }
}

// ----------------------------------------------------------------- out ----
// d_out layout: [out_next: q][inp: 12q][out_stdp: 12q]
__global__ __launch_bounds__(256) void tnn_out(float* __restrict__ out, int q,
                                               const float* __restrict__ wsF,
                                               const int* __restrict__ wsI) {
  int r = blockIdx.x * 256 + threadIdx.x;
  if (r >= q) return;
  float minv = wsF[36];
  int   iid  = wsI[38];
  float val  = (r == iid) ? minv : __builtin_inff();
  out[r] = val;                                            // out_next

  f4 t0 = {wsF[0], wsF[1], wsF[2],  wsF[3]};
  f4 t1 = {wsF[4], wsF[5], wsF[6],  wsF[7]};
  f4 t2 = {wsF[8], wsF[9], wsF[10], wsF[11]};
  f4* ip = (f4*)(out + (size_t)q + (size_t)r * 12);        // inp row
  __builtin_nontemporal_store(t0, ip + 0);
  __builtin_nontemporal_store(t1, ip + 1);
  __builtin_nontemporal_store(t2, ip + 2);

  f4 v4 = {val, val, val, val};
  f4* sp = (f4*)(out + (size_t)q * 13 + (size_t)r * 12);   // out_stdp row
  __builtin_nontemporal_store(v4, sp + 0);
  __builtin_nontemporal_store(v4, sp + 1);
  __builtin_nontemporal_store(v4, sp + 2);
}

// -------------------------------------------------------------- launch ----
extern "C" void kernel_launch(void* const* d_in, const int* in_sizes, int n_in,
                              void* d_out, int out_size, void* d_ws, size_t ws_size,
                              hipStream_t stream) {
  (void)n_in; (void)out_size; (void)ws_size;
  const float* data    = (const float*)d_in[0];
  const float* weights = (const float*)d_in[1];
  float* out = (float*)d_out;
  float*    wsF = (float*)d_ws;
  unsigned* wsU = (unsigned*)d_ws;
  int*      wsI = (int*)d_ws;

  const int q   = in_sizes[1] / P;        // 524288
  const int nb1 = q / (32 * TPW * 8);     // 8 waves/block, 32*TPW neurons/wave

  float* partEc  = wsF + 64;
  float* partPot = wsF + 64 + nb1;
  int*   partIdx = wsI + 64 + 2 * nb1;

  hipLaunchKernelGGL(tnn_prep, dim3(1), dim3(1), 0, stream, data, wsF, wsU);
  hipLaunchKernelGGL(tnn_ec, dim3(nb1), dim3(256), 0, stream,
                     weights, wsF, wsU, partEc, partPot, partIdx);
  hipLaunchKernelGGL(tnn_wta, dim3(1), dim3(256), 0, stream,
                     partEc, partPot, partIdx, nb1, wsF, wsI);
  hipLaunchKernelGGL(tnn_out, dim3((q + 255) / 256), dim3(256), 0, stream,
                     out, q, wsF, wsI);
}
